// TABL_6030134083634
// MI455X (gfx1250) — matrix operand order repaired and checked by reading, not compiled, not run on hardware
//
#include <hip/hip_runtime.h>
#include <hip/hip_bf16.h>

// ---------------------------------------------------------------------------
// Shapes
// ---------------------------------------------------------------------------
#define B_   8192
#define D_   40
#define T_   100
#define P_   120
#define Q_   100
#define KD   64            // D padded to K-multiple of 32 (bf16 WMMA K=32)
#define KT   128           // T padded to K-multiple of 32
#define TP   112           // T / Q padded to N-multiple of 16
#define PP   128           // P padded to M-multiple of 16
#define PT2  (PP * TP)     // 14336 padded (p,t) positions = 56 * 256 exactly

typedef __bf16 bf16;
typedef __attribute__((ext_vector_type(16))) __bf16 v16bf;
typedef __attribute__((ext_vector_type(8)))  __bf16 v8bf;
typedef __attribute__((ext_vector_type(8)))  float  v8f;

// ---------------------------------------------------------------------------
// WMMA fragment loaders (wave32, 16x16x32 bf16)
//
// A (16x32, MxK): lane L holds row m = L&15; element i lives in VGPR i/2 and
// k = 16*(L>>4) + i  -> two contiguous 16-byte runs per lane
// when A is stored row-major [m][k] in LDS.
//
// B (32x16, KxN): lane L holds column n = L&15; element i has
// k = 16*(L>>4) + i -> one contiguous 32-byte run per lane when B is staged
// TRANSPOSED in LDS as Bt[n][k].
// ---------------------------------------------------------------------------
__device__ __forceinline__ v16bf load_a_frag(const bf16* base, int ld) {
  const int lane = threadIdx.x & 31;
  const int row  = lane & 15;
  const int kh   = (lane >> 4) << 4;            // 0 or 16
  const bf16* p  = base + row * ld;
  v8bf lo = *(const v8bf*)(p + kh);             // k = kh .. kh+7
  v8bf hi = *(const v8bf*)(p + kh + 8);         // k = kh+8 .. kh+15
  return __builtin_shufflevector(lo, hi, 0,1,2,3,4,5,6,7,8,9,10,11,12,13,14,15);
}

__device__ __forceinline__ v16bf load_b_frag(const bf16* baseT, int ld) {
  const int lane = threadIdx.x & 31;
  const int n  = lane & 15;
  const int k0 = (lane >> 4) << 4;              // 0 or 16
  return *(const v16bf*)(baseT + n * ld + k0);
}

__device__ __forceinline__ v8f wmma_bf16(v16bf a, v16bf b, v8f c) {
  return __builtin_amdgcn_wmma_f32_16x16x32_bf16(
      /*neg_a=*/false, a, /*neg_b=*/false, b,
      /*c_mod=*/(short)0, c, /*reuse_a=*/false, /*reuse_b=*/false);
}

// ---------------------------------------------------------------------------
// Kernel 1: per-sample X1 = W1 @ X[b]   and   E = X1 @ W
// 256 threads = 8 waves; wave w owns output rows [16w, 16w+16).
// X1g / Eg are stored PADDED as [B][PP][TP]; padding elements are exactly 0
// (they are produced from zero-padded LDS operands), so the writeback is
// completely unconditional -> no exec-mask churn.
// ---------------------------------------------------------------------------
__global__ __launch_bounds__(256) void k1_x1_e(
    const float* __restrict__ X,  const float* __restrict__ W1,
    const float* __restrict__ W,  float* __restrict__ X1g,
    float* __restrict__ Eg)
{
  __shared__ __align__(32) bf16 sW1[PP * KD];   // A: W1 [m][d]      (16 KB)
  __shared__ __align__(32) bf16 sXt[TP * KD];   // Bt: X[b]^T [t][d] (14 KB)
  __shared__ __align__(32) bf16 sX1[PP * KT];   // A: X1 [m][t]      (32 KB)
  __shared__ __align__(32) bf16 sWt[TP * KT];   // Bt: W^T [s][t]    (28 KB)

  const int b   = blockIdx.x;
  const int tid = threadIdx.x;

  // zero-pad everything once
  for (int i = tid; i < PP * KD; i += 256) sW1[i] = (bf16)0.f;
  for (int i = tid; i < TP * KD; i += 256) sXt[i] = (bf16)0.f;
  for (int i = tid; i < PP * KT; i += 256) sX1[i] = (bf16)0.f;
  for (int i = tid; i < TP * KT; i += 256) sWt[i] = (bf16)0.f;
  __syncthreads();

  for (int i = tid; i < P_ * D_; i += 256) {            // W1 [P][D]
    int p = i / D_, d = i % D_;
    sW1[p * KD + d] = (bf16)W1[i];
  }
  for (int i = tid; i < D_ * T_; i += 256) {            // X[b] [D][T] -> [t][d]
    int d = i / T_, t = i % T_;
    sXt[t * KD + d] = (bf16)X[(size_t)b * (D_ * T_) + i];
  }
  for (int i = tid; i < T_ * T_; i += 256) {            // W [T][S] -> [s][t]
    int t = i / T_, s = i % T_;
    sWt[s * KT + t] = (bf16)W[i];
  }
  // prefetch next sample's X tile into cache (global_prefetch_b8 path)
  if (b + 1 < B_) {
    const char* nx = (const char*)(X + (size_t)(b + 1) * (D_ * T_));
    __builtin_prefetch(nx + tid * 64, 0, 1);
  }
  __syncthreads();

  const int w    = tid >> 5;
  const int lane = tid & 31;
  const int m0   = w * 16;
  const int cg   = lane >> 4;     // C/D: VGPR r -> m = m0 + r + 8*cg
  const int col  = lane & 15;     //            -> n = 16*ntile + col

  float* x1row = X1g + (size_t)b * PT2;

  // --- X1 = W1 @ X[b] : K = 64 (2 steps), N = 112 (7 tiles) ---
  for (int n = 0; n < 7; ++n) {
    v8f acc = {};
#pragma unroll
    for (int k = 0; k < 2; ++k) {
      v16bf af = load_a_frag(&sW1[m0 * KD + 32 * k], KD);
      v16bf bf = load_b_frag(&sXt[(n * 16) * KD + 32 * k], KD);
      acc = wmma_bf16(af, bf, acc);
    }
#pragma unroll
    for (int r = 0; r < 8; ++r) {
      int m = m0 + r + 8 * cg;           // 0..127, always in padded bounds
      int t = n * 16 + col;              // 0..111
      x1row[m * TP + t] = acc[r];        // f32 to (padded) workspace
      sX1[m * KT + t]   = (bf16)acc[r];  // bf16 A-operand for E
    }
  }
  __syncthreads();

  // --- E = X1 @ W : K = 128 (4 steps), N = 112 (7 tiles) ---
  float* erow = Eg + (size_t)b * PT2;
  for (int n = 0; n < 7; ++n) {
    v8f acc = {};
#pragma unroll
    for (int k = 0; k < 4; ++k) {
      v16bf af = load_a_frag(&sX1[m0 * KT + 32 * k], KT);
      v16bf bf = load_b_frag(&sWt[(n * 16) * KT + 32 * k], KT);
      acc = wmma_bf16(af, bf, acc);
    }
#pragma unroll
    for (int r = 0; r < 8; ++r) {
      int m = m0 + r + 8 * cg;
      int s = n * 16 + col;
      erow[m * TP + s] = acc[r];
    }
  }
}

// ---------------------------------------------------------------------------
// Kernel 2a: partial online max / sum-exp over the batch axis.
// grid (56 pt-chunks of 256 -> exactly PT2, 64 b-chunks of 128).
// Fully coalesced over pt, no bounds checks (padded columns are all 0.0).
// ---------------------------------------------------------------------------
__global__ __launch_bounds__(256) void k2_partial(
    const float* __restrict__ Eg, float* __restrict__ Mp, float* __restrict__ Sp)
{
  const int pt = blockIdx.x * 256 + threadIdx.x;     // < PT2 exactly
  const int y  = blockIdx.y;
  const float* colp = Eg + (size_t)(y * 128) * PT2 + pt;
  float m = -3.4e38f, s = 0.f;
  for (int j = 0; j < 128; ++j) {
    float e = colp[(size_t)j * PT2];
    if (e > m) { s = s * __expf(m - e) + 1.f; m = e; }
    else       { s += __expf(e - m); }
  }
  Mp[y * PT2 + pt] = m;
  Sp[y * PT2 + pt] = s;
}

// Kernel 2b: combine 64 partials -> final max / denom per padded (p,t)
__global__ __launch_bounds__(256) void k2_final(
    const float* __restrict__ Mp, const float* __restrict__ Sp,
    float* __restrict__ Mf, float* __restrict__ Sf)
{
  const int pt = blockIdx.x * 256 + threadIdx.x;     // < PT2 exactly
  float m = -3.4e38f, s = 0.f;
  for (int y = 0; y < 64; ++y) {
    float mm = Mp[y * PT2 + pt], ss = Sp[y * PT2 + pt];
    if (mm > m) { s = s * __expf(m - mm) + ss; m = mm; }
    else        { s += ss * __expf(mm - m); }
  }
  Mf[pt] = m;
  Sf[pt] = s;
}

// ---------------------------------------------------------------------------
// Kernel 3: per-sample  A = exp(E-M)/S ; tem = X1*(alpha*A + 1-alpha);
//           Y = tem @ W2 + bias.
// GEMM result is staged in LDS (f32) and written out with a flat coalesced
// loop fused with the bias add -> no divergent stores.
// ---------------------------------------------------------------------------
__global__ __launch_bounds__(256) void k3_out(
    const float* __restrict__ X1g, const float* __restrict__ Eg,
    const float* __restrict__ Mf,  const float* __restrict__ Sf,
    const float* __restrict__ alpha_p, const float* __restrict__ W2,
    const float* __restrict__ bias, float* __restrict__ Y)
{
  __shared__ __align__(32) bf16  sTem[PP * KT];   // A: tem [m][t]    (32 KB)
  __shared__ __align__(32) bf16  sW2t[TP * KT];   // Bt: W2^T [q][t]  (28 KB)
  __shared__ __align__(32) float sY[PP * TP];     // result staging (57.3 KB)

  const int b   = blockIdx.x;
  const int tid = threadIdx.x;
  const float alpha = alpha_p[0];

  for (int i = tid; i < PP * KT; i += 256) sTem[i] = (bf16)0.f;
  for (int i = tid; i < TP * KT; i += 256) sW2t[i] = (bf16)0.f;
  __syncthreads();

  for (int i = tid; i < T_ * Q_; i += 256) {           // W2 [T][Q] -> [q][t]
    int t = i / Q_, q = i % Q_;
    sW2t[q * KT + t] = (bf16)W2[i];
  }
  const float* e  = Eg  + (size_t)b * PT2;
  const float* x1 = X1g + (size_t)b * PT2;
  for (int i = tid; i < PT2; i += 256) {               // padded: no guards
    float a   = __expf(e[i] - Mf[i]) / Sf[i];
    float g   = alpha * a + (1.f - alpha);
    float tem = x1[i] * g;                             // padding: x1 = 0
    int p = i / TP, t = i % TP;
    sTem[p * KT + t] = (bf16)tem;
  }
  __syncthreads();

  const int w = tid >> 5, lane = tid & 31;
  const int m0 = w * 16, cg = lane >> 4, col = lane & 15;

  for (int n = 0; n < 7; ++n) {
    v8f acc = {};
#pragma unroll
    for (int k = 0; k < 4; ++k) {
      v16bf af = load_a_frag(&sTem[m0 * KT + 32 * k], KT);
      v16bf bf = load_b_frag(&sW2t[(n * 16) * KT + 32 * k], KT);
      acc = wmma_bf16(af, bf, acc);
    }
#pragma unroll
    for (int r = 0; r < 8; ++r) {
      int m = m0 + r + 8 * cg;
      int q = n * 16 + col;
      sY[m * TP + q] = acc[r];                         // unconditional
    }
  }
  __syncthreads();

  float* yrow = Y + (size_t)b * (P_ * Q_);
  for (int i = tid; i < P_ * Q_; i += 256) {           // coalesced writeout
    int p = i / Q_, q = i % Q_;
    yrow[i] = sY[p * TP + q] + bias[i];
  }
}

// ---------------------------------------------------------------------------
// Launch. Workspace layout (bytes):
//   X1 f32 [B][PP][TP]  469,762,048
//   E  f32 [B][PP][TP]  469,762,048
//   Mp/Sp  [64][PT2]    2 x 3,670,016
//   Mf/Sf  [PT2]        2 x 57,344
// total ~947 MB (device has 432 GiB)
// ---------------------------------------------------------------------------
extern "C" void kernel_launch(void* const* d_in, const int* in_sizes, int n_in,
                              void* d_out, int out_size, void* d_ws, size_t ws_size,
                              hipStream_t stream)
{
  const float* X     = (const float*)d_in[0];
  const float* W1    = (const float*)d_in[1];
  const float* W     = (const float*)d_in[2];
  const float* alpha = (const float*)d_in[3];
  const float* W2    = (const float*)d_in[4];
  const float* bias  = (const float*)d_in[5];
  float* Y = (float*)d_out;

  float* X1g = (float*)d_ws;
  float* Eg  = X1g + (size_t)B_ * PT2;
  float* Mp  = Eg  + (size_t)B_ * PT2;
  float* Sp  = Mp  + (size_t)64 * PT2;
  float* Mf  = Sp  + (size_t)64 * PT2;
  float* Sf  = Mf  + PT2;

  k1_x1_e<<<B_, 256, 0, stream>>>(X, W1, W, X1g, Eg);
  dim3 g2(56, 64);
  k2_partial<<<g2, 256, 0, stream>>>(Eg, Mp, Sp);
  k2_final<<<56, 256, 0, stream>>>(Mp, Sp, Mf, Sf);
  k3_out<<<B_, 256, 0, stream>>>(X1g, Eg, Mf, Sf, alpha, W2, bias, Y);
}
